// DenseCapsule_87351044866661
// MI455X (gfx1250) — matrix operand order repaired and checked by reading, not compiled, not run on hardware
//
#include <hip/hip_runtime.h>

// Problem sizes (fixed by the reference)
#define B_  32
#define I_  2048
#define D_  16
#define J_  64
#define C_  32
#define ITERS 5
#define EPS_ 1e-7f

typedef __bf16 v16bf __attribute__((ext_vector_type(16)));
typedef __bf16 v8bf  __attribute__((ext_vector_type(8)));
typedef float  v8f   __attribute__((ext_vector_type(8)));

__device__ __forceinline__ float bfu(unsigned s) {
    return __uint_as_float((s & 0xFFFFu) << 16);
}

// -----------------------------------------------------------------------------
// Kernel 1: u_hat[b,j,i,c] = sum_d x[b,i,d] * w[j,i,c,d], stored as bf16.
// Per i: GEMM M=32 (b), N=J*C=2048 (j,c), K=16 (d) via v_wmma_f32_16x16x32_bf16.
// Grid: (I_, 8); block 128 = 4 waves; each wave does 4 N-tiles of 16.
// -----------------------------------------------------------------------------
__global__ __launch_bounds__(128) void uhat_gemm_kernel(
    const float* __restrict__ x, const float* __restrict__ w,
    unsigned short* __restrict__ uhat)
{
    __shared__ float stage[4][32][16];

    const int wid   = threadIdx.x >> 5;
    const int lane  = threadIdx.x & 31;
    const int i     = blockIdx.x;
    const int nlo   = lane & 15;          // N index inside tile / M row for A
    const int half8 = (lane >> 4) * 8;    // 0 or 8
    const int d0    = half8;              // A: lanes 0-15 hold K=0..7, 16-31 hold K=8..15

    // ---- A fragments (16x32 bf16 layout, K=16..31 zero), shared by all N-tiles
    v16bf a0, a1;
    {
        const float4* xp0 = (const float4*)(x + ((size_t)nlo        * I_ + i) * D_ + d0);
        const float4* xp1 = (const float4*)(x + ((size_t)(nlo + 16) * I_ + i) * D_ + d0);
        float4 p0 = xp0[0], p1 = xp0[1];
        float4 q0 = xp1[0], q1 = xp1[1];
        a0[0]=(__bf16)p0.x; a0[1]=(__bf16)p0.y; a0[2]=(__bf16)p0.z; a0[3]=(__bf16)p0.w;
        a0[4]=(__bf16)p1.x; a0[5]=(__bf16)p1.y; a0[6]=(__bf16)p1.z; a0[7]=(__bf16)p1.w;
        a1[0]=(__bf16)q0.x; a1[1]=(__bf16)q0.y; a1[2]=(__bf16)q0.z; a1[3]=(__bf16)q0.w;
        a1[4]=(__bf16)q1.x; a1[5]=(__bf16)q1.y; a1[6]=(__bf16)q1.z; a1[7]=(__bf16)q1.w;
#pragma unroll
        for (int k = 8; k < 16; ++k) { a0[k] = (__bf16)0.0f; a1[k] = (__bf16)0.0f; }
    }

    const int nt0 = blockIdx.y * 16 + wid * 4;
#pragma unroll 1
    for (int t = 0; t < 4; ++t) {
        const int nt = nt0 + t;
        const int j  = nt >> 1;
        const int c0 = (nt & 1) << 4;

        // ---- B fragment: column n = lane%16; lanes 0-15 hold K=0..7, 16-31 hold K=8..15; elems 8-15 zero
        v16bf bfr;
        {
            const float4* wp = (const float4*)(w + (((size_t)j * I_ + i) * C_ + (c0 + nlo)) * D_ + half8);
            float4 w0 = wp[0], w1 = wp[1];
            bfr[0]=(__bf16)w0.x;  bfr[1]=(__bf16)w0.y;  bfr[2]=(__bf16)w0.z;  bfr[3]=(__bf16)w0.w;
            bfr[4]=(__bf16)w1.x;  bfr[5]=(__bf16)w1.y;  bfr[6]=(__bf16)w1.z;  bfr[7]=(__bf16)w1.w;
#pragma unroll
            for (int k = 8; k < 16; ++k) bfr[k] = (__bf16)0.0f;
        }
        if (t < 3 && lane < 16) {
            const int ntn = nt + 1;
            __builtin_prefetch(
                w + (((size_t)(ntn >> 1) * I_ + i) * C_ + (((ntn & 1) << 4) + lane)) * D_, 0, 0);
        }

        v8f acc0 = {0,0,0,0,0,0,0,0};
        v8f acc1 = {0,0,0,0,0,0,0,0};
        acc0 = __builtin_amdgcn_wmma_f32_16x16x32_bf16(false, a0, false, bfr,
                                                       (short)0, acc0, false, false);
        acc1 = __builtin_amdgcn_wmma_f32_16x16x32_bf16(false, a1, false, bfr,
                                                       (short)0, acc1, false, false);

        // ---- stage C tiles in LDS ([m=b][n=c]) so stores are one 32B row per lane
#pragma unroll
        for (int r = 0; r < 8; ++r) {
            stage[wid][half8 + r][nlo]      = acc0[r];   // m = 0..15
            stage[wid][16 + half8 + r][nlo] = acc1[r];   // m = 16..31
        }
        __syncthreads();
        {
            const float* row = stage[wid][lane];         // b = lane, 16 c-values
            v8bf o0, o1;
#pragma unroll
            for (int k = 0; k < 8; ++k) {                // lowers to v_cvt_pk_bf16_f32
                o0[k] = (__bf16)row[k];
                o1[k] = (__bf16)row[8 + k];
            }
            unsigned short* up = uhat + (((size_t)lane * J_ + j) * I_ + i) * C_ + c0;
            *(v8bf*)(up)     = o0;
            *(v8bf*)(up + 8) = o1;
        }
        __syncthreads();
    }
}

// -----------------------------------------------------------------------------
// Kernel 2: per (b,i): mx = max_j b_ij, rs = 1/sum_j exp(b_ij - mx)
// -----------------------------------------------------------------------------
__global__ __launch_bounds__(256) void softmax_stats_kernel(
    const float* __restrict__ bij, float* __restrict__ mx, float* __restrict__ rs)
{
    const int t = blockIdx.x * 256 + threadIdx.x;     // < B_*I_
    const int b = t >> 11;
    const int i = t & (I_ - 1);
    const float* p = bij + (size_t)b * J_ * I_ + i;
    float m = -3.402823466e38f;
#pragma unroll 8
    for (int j = 0; j < J_; ++j) m = fmaxf(m, p[(size_t)j * I_]);
    float s = 0.f;
#pragma unroll 8
    for (int j = 0; j < J_; ++j) s += __expf(p[(size_t)j * I_] - m);
    mx[t] = m;
    rs[t] = 1.0f / s;
}

// -----------------------------------------------------------------------------
// Kernel 3: s_j = sum_i softmax(b)_ij * u_hat[b,j,i,:], then squash -> v.
// One wave per (b,j). lane -> (i_sub = lane>>2, c-range = (lane&3)*8).
// -----------------------------------------------------------------------------
__global__ __launch_bounds__(256) void sj_squash_kernel(
    const unsigned short* __restrict__ uhat, const float* __restrict__ bij,
    const float* __restrict__ mx, const float* __restrict__ rs,
    float* __restrict__ vout)
{
    const int wid  = threadIdx.x >> 5;
    const int lane = threadIdx.x & 31;
    const int pair = blockIdx.x * 8 + wid;            // b*J_ + j
    const int b    = pair >> 6;
    const int isub = lane >> 2;
    const int c0   = (lane & 3) << 3;

    const unsigned short* up = uhat + (size_t)pair * I_ * C_;
    const float* bp = bij + (size_t)pair * I_;
    const float* mp = mx + (size_t)b * I_;
    const float* rp = rs + (size_t)b * I_;

    float acc[8] = {0,0,0,0,0,0,0,0};
#pragma unroll 2
    for (int s = 0; s < I_ / 8; ++s) {
        const int i = s * 8 + isub;
        const float cij = __expf(bp[i] - mp[i]) * rp[i];
        const uint4 pk = *(const uint4*)(up + (size_t)i * C_ + c0);
        acc[0] += cij * bfu(pk.x);       acc[1] += cij * bfu(pk.x >> 16);
        acc[2] += cij * bfu(pk.y);       acc[3] += cij * bfu(pk.y >> 16);
        acc[4] += cij * bfu(pk.z);       acc[5] += cij * bfu(pk.z >> 16);
        acc[6] += cij * bfu(pk.w);       acc[7] += cij * bfu(pk.w >> 16);
    }
#pragma unroll
    for (int k = 0; k < 8; ++k) {
        acc[k] += __shfl_xor(acc[k], 4, 32);
        acc[k] += __shfl_xor(acc[k], 8, 32);
        acc[k] += __shfl_xor(acc[k], 16, 32);
    }
    // squash (reference order: add EPS elementwise BEFORE norm)
    float vk[8]; float nrm = 0.f;
#pragma unroll
    for (int k = 0; k < 8; ++k) { vk[k] = acc[k] + EPS_; nrm += vk[k] * vk[k]; }
    nrm += __shfl_xor(nrm, 1, 32);
    nrm += __shfl_xor(nrm, 2, 32);
    const float scale = nrm / (1.0f + nrm) * rsqrtf(nrm);
    if (lane < 4) {
        float* op = vout + (size_t)pair * C_ + c0;
        *(float4*)(op)     = make_float4(vk[0]*scale, vk[1]*scale, vk[2]*scale, vk[3]*scale);
        *(float4*)(op + 4) = make_float4(vk[4]*scale, vk[5]*scale, vk[6]*scale, vk[7]*scale);
    }
}

// -----------------------------------------------------------------------------
// Kernel 4: b_ij += sum_c v[b,j,c] * u_hat[b,j,i,c]. One wave per (b,j),
// lane -> i; each lane reads a contiguous 64B row of u_hat.
// -----------------------------------------------------------------------------
__global__ __launch_bounds__(256) void bupdate_kernel(
    const unsigned short* __restrict__ uhat, const float* __restrict__ v,
    float* __restrict__ bij)
{
    const int wid  = threadIdx.x >> 5;
    const int lane = threadIdx.x & 31;
    const int pair = blockIdx.x * 8 + wid;

    const unsigned short* up = uhat + (size_t)pair * I_ * C_;
    float* bp = bij + (size_t)pair * I_;

    float vv[32];
    const float4* vp = (const float4*)(v + (size_t)pair * C_);
#pragma unroll
    for (int q = 0; q < 8; ++q) {
        float4 t = vp[q];
        vv[q*4+0] = t.x; vv[q*4+1] = t.y; vv[q*4+2] = t.z; vv[q*4+3] = t.w;
    }
#pragma unroll 1
    for (int s = 0; s < I_ / 32; ++s) {
        const int i = s * 32 + lane;
        const unsigned short* u = up + (size_t)i * C_;
        float dot = 0.f;
#pragma unroll
        for (int q = 0; q < 4; ++q) {
            uint4 pk = *(const uint4*)(u + q * 8);
            dot += vv[q*8+0] * bfu(pk.x) + vv[q*8+1] * bfu(pk.x >> 16);
            dot += vv[q*8+2] * bfu(pk.y) + vv[q*8+3] * bfu(pk.y >> 16);
            dot += vv[q*8+4] * bfu(pk.z) + vv[q*8+5] * bfu(pk.z >> 16);
            dot += vv[q*8+6] * bfu(pk.w) + vv[q*8+7] * bfu(pk.w >> 16);
        }
        bp[i] += dot;
    }
}

__global__ __launch_bounds__(256) void zero_kernel(float* __restrict__ p, int n)
{
    const int t = blockIdx.x * 256 + threadIdx.x;
    if (t < n) p[t] = 0.f;
}

// -----------------------------------------------------------------------------
extern "C" void kernel_launch(void* const* d_in, const int* in_sizes, int n_in,
                              void* d_out, int out_size, void* d_ws, size_t ws_size,
                              hipStream_t stream)
{
    const float* x = (const float*)d_in[0];   // [B, I, D]
    const float* w = (const float*)d_in[1];   // [J, I, C, D]
    float* out = (float*)d_out;               // [B, J, C]
    char* ws = (char*)d_ws;

    size_t off = 0;
    unsigned short* uhat = (unsigned short*)(ws + off);  off += (size_t)B_*J_*I_*C_*sizeof(unsigned short); // 268 MB
    float* bij = (float*)(ws + off);                     off += (size_t)B_*J_*I_*sizeof(float);             // 16.8 MB
    float* mx  = (float*)(ws + off);                     off += (size_t)B_*I_*sizeof(float);
    float* rs  = (float*)(ws + off);                     off += (size_t)B_*I_*sizeof(float);
    float* v   = (float*)(ws + off);                     off += (size_t)B_*J_*C_*sizeof(float);

    zero_kernel<<<(B_*J_*I_) / 256, 256, 0, stream>>>(bij, B_*J_*I_);
    uhat_gemm_kernel<<<dim3(I_, (J_*C_) / 16 / 16), 128, 0, stream>>>(x, w, uhat);

    for (int it = 0; it < ITERS; ++it) {
        softmax_stats_kernel<<<(B_*I_) / 256, 256, 0, stream>>>(bij, mx, rs);
        float* vdst = (it == ITERS - 1) ? out : v;
        sj_squash_kernel<<<(B_*J_) / 8, 256, 0, stream>>>(uhat, bij, mx, rs, vdst);
        if (it < ITERS - 1)
            bupdate_kernel<<<(B_*J_) / 8, 256, 0, stream>>>(uhat, v, bij);
    }
}
